// Model_exp_79182017069103
// MI455X (gfx1250) — matrix operand order repaired and checked by reading, not compiled, not run on hardware
//
#include <hip/hip_runtime.h>
#include <cstddef>

// ---------------- problem constants ----------------
#define BB   8192
#define MM   64
#define UU   128
#define TPP  128
#define FDD  50
#define RR   4
#define NBB  16      // batch rows per workgroup (== WMMA M tile)
#define NWG  (BB / NBB)

// ---------------- ws layout (f16 halves, then f32) ----------------
#define OFF_W1T  0         // [128][128]  gate3_w rows   0..127 (h_pre part), col-major-by-output
#define OFF_W2T  16384     // [128][128]  gate3_w rows 128..255 (interact part)
#define OFF_WT   32768     // [128][256]  Wt_g^T
#define OFF_WA   65536     // [128][256]  Wa_g^T
#define OFF_WH   98304     // [128][256]  Wh_g^T
#define OFF_OUT  131072    // [128][256]  out_w^T
#define OFF_FT   163840    // [512][128]  time_weight[r][k][u] -> row o=r*128+u
#define OFF_FA   229376
#define OFF_FH   294912
#define F16_TOTAL 360448
#define F16_BYTES (F16_TOTAL * 2)
// f32 section (float offsets inside wsf)
#define WSF_CT   0         // [128] column sums of time  factor rows of gate3_w
#define WSF_CA   128
#define WSF_CH   256
#define WSF_FTB  384       // [512] time_weight[r][128][u]  (the "+1 pad row" bias)
#define WSF_FAB  896
#define WSF_FHB  1408

typedef __attribute__((ext_vector_type(16))) _Float16 v16h;
typedef __attribute__((ext_vector_type(8)))  _Float16 v8h;
typedef __attribute__((ext_vector_type(8)))  float    v8f;
typedef __attribute__((ext_vector_type(4)))  float    v4f;
typedef __attribute__((ext_vector_type(4)))  unsigned int v4u;

// ---------------- helpers ----------------
__device__ __forceinline__ v16h cat8(v8h lo, v8h hi) {
  v16h r;
#pragma unroll
  for (int i = 0; i < 8; ++i) { r[i] = lo[i]; r[i + 8] = hi[i]; }
  return r;
}

__device__ __forceinline__ v8h cvt8(const float* p) {
  v4f a = *(const v4f*)p;
  v4f b = *(const v4f*)(p + 4);
  v8h r;
#pragma unroll
  for (int i = 0; i < 4; ++i) { r[i] = (_Float16)a[i]; r[i + 4] = (_Float16)b[i]; }
  return r;
}

// A fragment (16x32 f16) from LDS row-major [16][ld]:
// lane l: row m = l&15, half-group g = l>>4; elems 0..7 -> K = 32c+8g+i, 8..15 -> K = 32c+16+8g+i
__device__ __forceinline__ v16h loadA_lds(const _Float16* X, int ld, int c, int lane) {
  int m = lane & 15, g = lane >> 4;
  const _Float16* p = X + m * ld + c * 32 + g * 8;
  v8h lo = *(const v8h*)p;
  v8h hi = *(const v8h*)(p + 16);
  return cat8(lo, hi);
}

// A fragment from global f32 row-major [rows][128], converted to f16 on the fly
__device__ __forceinline__ v16h loadA_g32(const float* base, int m0, int c, int lane) {
  int m = m0 + (lane & 15), g = lane >> 4;
  const float* p = base + m * 128 + c * 32 + g * 8;
  return cat8(cvt8(p), cvt8(p + 16));
}

// B fragment (32x16 f16) from col-major BT[u][ldk] (global or LDS):
// lane l: col u = col_base + (l&15); elems 0..7 -> K = 32c+8*(l>>4)+i, 8..15 -> K = 32c+16+8*(l>>4)+i
__device__ __forceinline__ v16h loadB(const _Float16* BT, int ldk, int col_base, int c, int lane) {
  int u = col_base + (lane & 15);
  int k = c * 32 + ((lane >> 4) << 3);
  const _Float16* p = BT + u * ldk + k;
  v8h lo = *(const v8h*)p;
  v8h hi = *(const v8h*)(p + 16);
  return cat8(lo, hi);
}

__device__ __forceinline__ v8f wmma_f16(v16h a, v16h b, v8f c) {
  return __builtin_amdgcn_wmma_f32_16x16x32_f16(false, a, false, b, (short)0, c, false, false);
}

__device__ __forceinline__ float sigm(float x) { return 1.0f / (1.0f + __expf(-x)); }

// ---------------- prep: f16 transposed weights ----------------
__global__ void prep_w(const float* Wt_g, const float* Wa_g, const float* Wh_g,
                       const float* tw, const float* aw, const float* hw,
                       const float* g3w, const float* out_w, _Float16* wsh) {
  int idx = blockIdx.x * blockDim.x + threadIdx.x;
  if (idx >= F16_TOTAL) return;
  float v;
  if (idx < 16384) {                    // W1T
    int u = idx >> 7, k = idx & 127;  v = g3w[k * 128 + u];
  } else if (idx < 32768) {             // W2T
    int l = idx - 16384; int u = l >> 7, k = l & 127;  v = g3w[(128 + k) * 128 + u];
  } else if (idx < 65536) {             // WtT
    int l = idx - 32768; int u = l >> 8, k = l & 255;  v = Wt_g[k * 128 + u];
  } else if (idx < 98304) {             // WaT
    int l = idx - 65536; int u = l >> 8, k = l & 255;  v = Wa_g[k * 128 + u];
  } else if (idx < 131072) {            // WhT
    int l = idx - 98304; int u = l >> 8, k = l & 255;  v = Wh_g[k * 128 + u];
  } else if (idx < 163840) {            // OutT
    int l = idx - 131072; int u = l >> 8, k = l & 255; v = out_w[k * 128 + u];
  } else if (idx < 229376) {            // FtT
    int l = idx - 163840; int o = l >> 7, k = l & 127; int r = o >> 7, u = o & 127;
    v = tw[(r * 129 + k) * 128 + u];
  } else if (idx < 294912) {            // FaT
    int l = idx - 229376; int o = l >> 7, k = l & 127; int r = o >> 7, u = o & 127;
    v = aw[(r * 129 + k) * 128 + u];
  } else {                              // FhT
    int l = idx - 294912; int o = l >> 7, k = l & 127; int r = o >> 7, u = o & 127;
    v = hw[(r * 129 + k) * 128 + u];
  }
  wsh[idx] = (_Float16)v;
}

// ---------------- prep: f32 reductions (factor column sums, fusion pad row) ----------------
__global__ void prep_s(const float* g3w, const float* tw, const float* aw, const float* hw,
                       float* wsf) {
  int t = blockIdx.x * blockDim.x + threadIdx.x;
  if (t < 384) {
    int u = t & 127, grp = t >> 7;          // 0=time,1=attempt,2=hint
    int r0 = 2 * UU + grp * FDD;
    float s = 0.f;
    for (int i = 0; i < FDD; ++i) s += g3w[(r0 + i) * 128 + u];
    wsf[grp * 128 + u] = s;
  } else if (t < 1920) {
    int l = t - 384; int grp = l / 512; int o = l % 512;
    int r = o >> 7, u = o & 127;
    const float* w = (grp == 0) ? tw : (grp == 1) ? aw : hw;
    wsf[384 + grp * 512 + o] = w[(r * 129 + 128) * 128 + u];
  }
}

// ---------------- fused main kernel: 16 batch rows / workgroup ----------------
__global__ __launch_bounds__(256) void fused_kt(
    const float* __restrict__ interact, const float* __restrict__ corr,
    const float* __restrict__ topic,    const float* __restrict__ timef,
    const float* __restrict__ attemptf, const float* __restrict__ hintf,
    const float* __restrict__ h_pre,
    const float* __restrict__ bt_g, const float* __restrict__ ba_g,
    const float* __restrict__ bh_g, const float* __restrict__ Wf,
    const float* __restrict__ bias, const float* __restrict__ g3b,
    const float* __restrict__ out_b,
    const _Float16* __restrict__ wsh, const float* __restrict__ wsf,
    float* __restrict__ out) {
  __shared__ __align__(16) _Float16 Xg[16 * 256];      // [h_tilde | interact] f16
  __shared__ __align__(16) _Float16 Xt[16 * 128];      // topic f16
  __shared__ __align__(16) _Float16 G3[3][16 * 128];   // time/attempt/hint gains f16
  __shared__ __align__(16) _Float16 W1Tl[128 * 128];   // LDS copy of W1T (reused 16x in Phase C)
  __shared__ float R5[5][16 * 128];                    // gate_t/a/h pre-act, out pre-act, W2 part
  __shared__ float Base[16 * 128];
  __shared__ float Lg[16 * 128];
  __shared__ float Cw[16 * 64];
  __shared__ float Fac[16 * 4];
  __shared__ float Pp[16 * 16];

  const int t    = threadIdx.x;
  const int lane = t & 31;
  // readfirstlane: provably wave-uniform -> scalar branches in B1 (EXEC stays all-1s for WMMA)
  const int wave = __builtin_amdgcn_readfirstlane(t >> 5);
  const int bbase = blockIdx.x * NBB;

  // ---------------- Phase A: h_tilde + f16 staging + W1T -> LDS ----------------
  {
    const v4u* src = (const v4u*)(wsh + OFF_W1T);
    v4u* dst = (v4u*)W1Tl;
    for (int i = t; i < 2048; i += 256) dst[i] = src[i];
  }
  for (int i = t; i < 16 * 64; i += 256) Cw[i] = corr[(size_t)bbase * MM + i];
  if (t < 16) {
    Fac[t * 4 + 0] = timef[bbase + t];
    Fac[t * 4 + 1] = attemptf[bbase + t];
    Fac[t * 4 + 2] = hintf[bbase + t];
  }
  {
    const int bl = t >> 4, u0 = (t & 15) * 8;
    const float* hp  = h_pre + (size_t)(bbase + bl) * MM * UU;
    const float* cwp = corr  + (size_t)(bbase + bl) * MM;
    float acc[8];
#pragma unroll
    for (int i = 0; i < 8; ++i) acc[i] = 0.f;
    for (int m = 0; m < MM; ++m) {
      float w = cwp[m];
      v4f x0 = *(const v4f*)(hp + m * 128 + u0);
      v4f x1 = *(const v4f*)(hp + m * 128 + u0 + 4);
#pragma unroll
      for (int i = 0; i < 4; ++i) { acc[i] += w * x0[i]; acc[4 + i] += w * x1[i]; }
    }
#pragma unroll
    for (int i = 0; i < 8; ++i) {
      Xg[bl * 256 + u0 + i]       = (_Float16)acc[i];
      Xg[bl * 256 + 128 + u0 + i] = (_Float16)interact[(size_t)(bbase + bl) * UU + u0 + i];
      Xt[bl * 128 + u0 + i]       = (_Float16)topic[(size_t)(bbase + bl) * TPP + u0 + i];
    }
  }
  __syncthreads();

  // ---------------- Phase B1: gates / preds head / W2  (40 tiles, 5 per wave) ----------------
  for (int it = 0; it < 5; ++it) {
    const int id = wave + it * 8;     // wave-uniform -> scalar branch
    v8f acc = {};
    int ridx, n;
    if (id < 24) {
      const int tt = id >> 3; n = id & 7;
      const _Float16* BT = wsh + OFF_WT + tt * 32768;
#pragma unroll
      for (int c = 0; c < 8; ++c)
        acc = wmma_f16(loadA_lds(Xg, 256, c, lane), loadB(BT, 256, n * 16, c, lane), acc);
      ridx = tt;
    } else if (id < 32) {
      n = id - 24;
      const _Float16* BT = wsh + OFF_OUT;
#pragma unroll
      for (int c = 0; c < 8; ++c) {
        v16h a = (c < 4) ? loadA_lds(Xg, 256, c, lane) : loadA_lds(Xt, 128, c - 4, lane);
        acc = wmma_f16(a, loadB(BT, 256, n * 16, c, lane), acc);
      }
      ridx = 3;
    } else {
      n = id - 32;
      const _Float16* BT = wsh + OFF_W2T;
#pragma unroll
      for (int c = 0; c < 4; ++c)
        acc = wmma_f16(loadA_lds(Xg, 256, c + 4, lane), loadB(BT, 128, n * 16, c, lane), acc);
      ridx = 4;
    }
    float* Rb = &R5[ridx][0];
    const int col = n * 16 + (lane & 15);
    const int rb  = (lane >> 4) * 8;
#pragma unroll
    for (int r2 = 0; r2 < 8; ++r2) Rb[(r2 + rb) * 128 + col] = acc[r2];
  }
  __syncthreads();

  // ---------------- elementwise: gains (f16), base, preds partials ----------------
  {
    const int bl = t >> 4, u0 = (t & 15) * 8;
    const float tf = Fac[bl * 4 + 0], af = Fac[bl * 4 + 1], hf = Fac[bl * 4 + 2];
    const float gt = 0.3f + 0.7f / (1.f + __expf(-10.f * (tf - 0.3f)));
    const float ga = 0.3f + 0.7f / (1.f + __expf(-10.f * (af - 0.3f)));
    const float gh = 0.3f + 0.7f / (1.f + __expf(-10.f * (hf - 0.3f)));
    float ps = 0.f;
#pragma unroll
    for (int i = 0; i < 8; ++i) {
      const int u = u0 + i, ix = bl * 128 + u;
      G3[0][ix] = (_Float16)sigm((R5[0][ix] + bt_g[u]) * gt);
      G3[1][ix] = (_Float16)sigm((R5[1][ix] + ba_g[u]) * ga);
      G3[2][ix] = (_Float16)sigm((R5[2][ix] + bh_g[u]) * gh);
      Base[ix] = R5[4][ix] + wsf[WSF_CT + u] * tf + wsf[WSF_CA + u] * af +
                 wsf[WSF_CH + u] * hf + g3b[u];
      ps += sigm(R5[3][ix] + out_b[u]);
    }
    Pp[t] = ps;
  }
  __syncthreads();
  if (t < 16) {
    float s = 0.f;
    for (int i = 0; i < 16; ++i) s += Pp[t * 16 + i];
    out[bbase + t] = s * (1.0f / (float)UU);
  }

  // ---------------- Phase B2: rank-R fusion (wave w owns u-tile w) ----------------
  {
    v16h At[4], Aa[4], Ah[4];
#pragma unroll
    for (int c = 0; c < 4; ++c) {
      At[c] = loadA_lds(&G3[0][0], 128, c, lane);
      Aa[c] = loadA_lds(&G3[1][0], 128, c, lane);
      Ah[c] = loadA_lds(&G3[2][0], 128, c, lane);
    }
    const int ul = wave * 16 + (lane & 15);
    v8f acc = {};
#pragma unroll
    for (int r4 = 0; r4 < 4; ++r4) {
      v8f ct8 = {}, ca8 = {}, ch8 = {};
      const int colb = r4 * 128 + wave * 16;
#pragma unroll
      for (int c = 0; c < 4; ++c) {
        ct8 = wmma_f16(At[c], loadB(wsh + OFF_FT, 128, colb, c, lane), ct8);
        ca8 = wmma_f16(Aa[c], loadB(wsh + OFF_FA, 128, colb, c, lane), ca8);
        ch8 = wmma_f16(Ah[c], loadB(wsh + OFF_FH, 128, colb, c, lane), ch8);
      }
      const int o = r4 * 128 + ul;
      const float ftb = wsf[WSF_FTB + o], fab = wsf[WSF_FAB + o], fhb = wsf[WSF_FHB + o];
      const float wf = Wf[r4];
#pragma unroll
      for (int k = 0; k < 8; ++k)
        acc[k] += wf * (ct8[k] + ftb) * (ca8[k] + fab) * (ch8[k] + fhb);
    }
    const float bi = bias[ul];
    const int rb = (lane >> 4) * 8;
#pragma unroll
    for (int k = 0; k < 8; ++k) {
      float v = acc[k] + bi;
      Lg[(k + rb) * 128 + ul] = v > 0.f ? v : 0.f;
    }
  }
  __syncthreads();

  // ---------------- Phase C: per-b  h_pre @ W1 + fused epilogue ----------------
  for (int bl = 0; bl < NBB; ++bl) {
    const float* hp = h_pre + (size_t)(bbase + bl) * MM * UU;
    if (bl + 1 < NBB) {   // prefetch next tile (global_prefetch_b8), L2-warm -> WGP caches
      const float* hpn = h_pre + (size_t)(bbase + bl + 1) * MM * UU;
      __builtin_prefetch(hpn + t * 32, 0, 3);
    }
    const int ms = (wave >> 1) * 16;              // m-strip
    v16h A[4];
#pragma unroll
    for (int c = 0; c < 4; ++c) A[c] = loadA_g32(hp, ms, c, lane);
#pragma unroll
    for (int j = 0; j < 4; ++j) {
      const int n0 = ((wave & 1) * 4 + j) * 16;
      v8f acc = {};
#pragma unroll
      for (int c = 0; c < 4; ++c)
        acc = wmma_f16(A[c], loadB(W1Tl, 128, n0, c, lane), acc);
      const int u = n0 + (lane & 15);
      const float bs = Base[bl * 128 + u];
      const float lg = Lg[bl * 128 + u];
      float* op = out + BB + (size_t)(bbase + bl) * MM * UU + u;
      const int rb = (lane >> 4) * 8;
#pragma unroll
      for (int k = 0; k < 8; ++k) {
        const int m = ms + k + rb;
        const float sg = sigm(acc[k] + bs);
        op[(size_t)m * 128] = hp[m * 128 + u] * sg + Cw[bl * 64 + m] * lg;
      }
    }
  }
}

// ---------------- host entry ----------------
extern "C" void kernel_launch(void* const* d_in, const int* in_sizes, int n_in,
                              void* d_out, int out_size, void* d_ws, size_t ws_size,
                              hipStream_t stream) {
  (void)in_sizes; (void)n_in; (void)out_size; (void)ws_size;
  const float* interact = (const float*)d_in[0];
  const float* corr     = (const float*)d_in[1];
  const float* topic    = (const float*)d_in[2];
  const float* timef    = (const float*)d_in[3];
  const float* attemptf = (const float*)d_in[4];
  const float* hintf    = (const float*)d_in[5];
  const float* h_pre    = (const float*)d_in[6];
  const float* Wt_g     = (const float*)d_in[7];
  const float* bt_g     = (const float*)d_in[8];
  const float* Wa_g     = (const float*)d_in[9];
  const float* ba_g     = (const float*)d_in[10];
  const float* Wh_g     = (const float*)d_in[11];
  const float* bh_g     = (const float*)d_in[12];
  const float* tw       = (const float*)d_in[13];
  const float* aw       = (const float*)d_in[14];
  const float* hw       = (const float*)d_in[15];
  const float* Wf       = (const float*)d_in[16];
  const float* bias     = (const float*)d_in[17];
  const float* g3w      = (const float*)d_in[18];
  const float* g3b      = (const float*)d_in[19];
  const float* out_w    = (const float*)d_in[20];
  const float* out_b    = (const float*)d_in[21];
  float*     out = (float*)d_out;
  _Float16*  wsh = (_Float16*)d_ws;
  float*     wsf = (float*)((char*)d_ws + F16_BYTES);

  prep_w<<<(F16_TOTAL + 255) / 256, 256, 0, stream>>>(Wt_g, Wa_g, Wh_g, tw, aw, hw, g3w, out_w, wsh);
  prep_s<<<8, 256, 0, stream>>>(g3w, tw, aw, hw, wsf);
  fused_kt<<<NWG, 256, 0, stream>>>(interact, corr, topic, timef, attemptf, hintf, h_pre,
                                    bt_g, ba_g, bh_g, Wf, bias, g3b, out_b, wsh, wsf, out);
}
